// AxialSelfAttention_81956565942913
// MI455X (gfx1250) — matrix operand order repaired and checked by reading, not compiled, not run on hardware
//
#include <hip/hip_runtime.h>
#include <hip/hip_bf16.h>

#define E_DIM 512
#define N_TOK 2048
#define BATCH 32
#define QSCALE 0.04419417382415922f /* 1/sqrt(512) */

typedef __attribute__((ext_vector_type(16))) _Float16 v16h;
typedef __attribute__((ext_vector_type(8)))  _Float16 v8h;
typedef __attribute__((ext_vector_type(8)))  float    v8f;
typedef unsigned int u32x4 __attribute__((ext_vector_type(4)));
typedef int          i32x8 __attribute__((ext_vector_type(8)));
typedef int          i32x4 __attribute__((ext_vector_type(4)));

static __device__ __forceinline__ v8f wmma16(v16h a, v16h b, v8f c) {
  return __builtin_amdgcn_wmma_f32_16x16x32_f16(false, a, false, b, (short)0, c,
                                                false, false);
}

// A-fragment (16x32 f16): lane holds row (lane&15); K = (lane>>4)*16 + 0..15
static __device__ __forceinline__ v16h load_a_h(const _Float16* t, int stride, int lane) {
  const int r = lane & 15;
  const int hb = (lane >> 4) << 4;
  const _Float16* p = t + r * stride + hb;
  union { v16h v; v8h h[2]; } u;
  u.h[0] = *(const v8h*)(p);
  u.h[1] = *(const v8h*)(p + 8);
  return u.v;
}

// B-fragment (32x16 f16): lane holds column (lane&15); K = (lane>>4)*16 + 0..15
static __device__ __forceinline__ v16h load_b_h(const _Float16* t, int stride, int lane) {
  const int c = lane & 15;
  const int kb = (lane >> 4) << 4;
  return *(const v16h*)(t + c * stride + kb);
}

// A-fragment sourced from f32 memory with on-the-fly f16 conversion
static __device__ __forceinline__ v16h load_a_f32(const float* t, int stride, int lane) {
  const int r = lane & 15;
  const int hb = (lane >> 4) << 4;
  const float* p = t + r * stride + hb;
  v8f f0 = *(const v8f*)(p);
  v8f f1 = *(const v8f*)(p + 8);
  v16h o;
#pragma unroll
  for (int i = 0; i < 8; ++i) { o[i] = (_Float16)f0[i]; o[i + 8] = (_Float16)f1[i]; }
  return o;
}

// ---------------------------------------------------------------------------
// Tensor Data Mover: async 2D tile load global -> LDS.
// Descriptor bit layout per CDNA5 ISA (D# groups 0/1).
// data_size = 2 bytes (f16). Dims/strides in element units.
// 6-arg builtin variant (clang-23): groups 2/3 + trailing i32x8 are unused
// (NULL) for 2D tiles -> zero-filled.
// ---------------------------------------------------------------------------
static __device__ __forceinline__ void tdm_load_2d_f16(
    unsigned lds_off, const void* gaddr, unsigned tile_d0, unsigned tile_d1,
    unsigned tensor_d0, unsigned tensor_d1, unsigned stride0) {
  const unsigned long long ga = (unsigned long long)gaddr;
  u32x4 g0;
  g0[0] = 1u;                                   // count=1, user mode
  g0[1] = lds_off;                              // lds_addr (byte offset)
  g0[2] = (unsigned)(ga & 0xffffffffu);         // global_addr[31:0]
  g0[3] = (unsigned)((ga >> 32) & 0x1ffffffu)   // global_addr[56:32]
          | (2u << 30);                         // type = 2 ("image")
  i32x8 g1;
  g1[0] = (int)(1u << 16);                      // wg_mask=0, data_size=1 (2B)
  g1[1] = (int)((tensor_d0 & 0xffffu) << 16);   // tensor_dim0[15:0]
  g1[2] = (int)(((tensor_d0 >> 16) & 0xffffu) | ((tensor_d1 & 0xffffu) << 16));
  g1[3] = (int)(((tensor_d1 >> 16) & 0xffffu) | ((tile_d0 & 0xffffu) << 16));
  g1[4] = (int)(tile_d1 & 0xffffu);             // tile_dim1, tile_dim2=0
  g1[5] = (int)stride0;                         // tensor_dim0_stride[31:0]
  g1[6] = 0;                                    // stride0 hi / stride1 lo
  g1[7] = 0;
  const i32x4 z4 = {0, 0, 0, 0};
  const i32x8 z8 = {0, 0, 0, 0, 0, 0, 0, 0};
  __builtin_amdgcn_tensor_load_to_lds(g0, g1, z4, z4, z8, 0);
}

// ---------------------------------------------------------------------------
// Kernel 1: qkv = x @ W + b  -> q (scaled by 1/sqrt(E)), k, v as f16
// ---------------------------------------------------------------------------
__global__ __launch_bounds__(256) void qkv_gemm_kernel(
    const float* __restrict__ x, const float* __restrict__ w,
    const float* __restrict__ bias, _Float16* __restrict__ qh,
    _Float16* __restrict__ kh, _Float16* __restrict__ vh) {
  __shared__ alignas(32) _Float16 sWt[64 * 32]; // [colLocal][k]
  const int lane = threadIdx.x & 31;
  const int wv = threadIdx.x >> 5;
  const int rowBase = blockIdx.x * 128 + wv * 16;
  const int colBase = blockIdx.y * 64;

  v8f acc[4] = {};
  for (int k0 = 0; k0 < E_DIM; k0 += 32) {
    {
      const int kl = threadIdx.x >> 3;
      const int c0 = (threadIdx.x & 7) * 8;
      v8f wd = *(const v8f*)(w + (size_t)(k0 + kl) * (3 * E_DIM) + colBase + c0);
#pragma unroll
      for (int j = 0; j < 8; ++j) sWt[(c0 + j) * 32 + kl] = (_Float16)wd[j];
    }
    __syncthreads();
    v16h a = load_a_f32(x + (size_t)rowBase * E_DIM + k0, E_DIM, lane);
#pragma unroll
    for (int ct = 0; ct < 4; ++ct) {
      v16h bb = load_b_h(sWt + ct * 16 * 32, 32, lane);
      acc[ct] = wmma16(a, bb, acc[ct]);
    }
    __syncthreads();
  }
#pragma unroll
  for (int ct = 0; ct < 4; ++ct) {
    const int gcol = colBase + ct * 16 + (lane & 15);
    const float bs = bias[gcol];
    _Float16* dst; int dcol; float sc = 1.0f;
    if (gcol < E_DIM)           { dst = qh; dcol = gcol;             sc = QSCALE; }
    else if (gcol < 2 * E_DIM)  { dst = kh; dcol = gcol - E_DIM;     }
    else                        { dst = vh; dcol = gcol - 2 * E_DIM; }
#pragma unroll
    for (int i = 0; i < 8; ++i) {
      const int grow = rowBase + i + ((lane & 16) ? 8 : 0);
      dst[(size_t)grow * E_DIM + dcol] = (_Float16)((acc[ct][i] + bs) * sc);
    }
  }
}

// ---------------------------------------------------------------------------
// Kernel 2: row attention, flash-style. Block = 8 waves, 64 query rows, full
// E=512 in registers. q fragments hoisted to registers (16 x v16h). Key chunk
// staged to LDS by the Tensor Data Mover; V^T staged in two 256-wide passes
// (interleaved e-tile ownership keeps all waves busy each pass).
// ---------------------------------------------------------------------------
__global__ __launch_bounds__(256) void row_attn_kernel(
    const _Float16* __restrict__ qh, const _Float16* __restrict__ kh,
    const _Float16* __restrict__ vh, float* __restrict__ out) {
  __shared__ alignas(32) _Float16 sK[32 * E_DIM];  // TDM-staged keys [kv][d]
  __shared__ alignas(32) float    sS[64 * 32];     // score chunk
  __shared__ alignas(32) _Float16 sP[64 * 32];     // probs (f16)
  __shared__ alignas(32) _Float16 sVt[256 * 32];   // V^T half-pass: [eLocal][kv]
  __shared__ float sFac[64];
  __shared__ float sInvL[64];

  const int lane = threadIdx.x & 31;
  const int wv = threadIdx.x >> 5;
  const int rowTile = wv & 3;     // 16-row q tile
  const int colPar = wv >> 2;     // e-tile parity (and key half for scores)
  const int b = blockIdx.y;
  const int q0 = blockIdx.x * 64;
  const _Float16* qb = qh + (size_t)b * N_TOK * E_DIM;
  const _Float16* kbp = kh + (size_t)b * N_TOK * E_DIM;
  const _Float16* vb = vh + (size_t)b * N_TOK * E_DIM;
  const unsigned sK_off = (unsigned)(unsigned long long)(void*)&sK[0];

  // hoist q A-fragments (rows fixed per wave) for the whole kernel
  v16h qf[16];
#pragma unroll
  for (int j = 0; j < 16; ++j)
    qf[j] = load_a_h(qb + (size_t)(q0 + rowTile * 16) * E_DIM + j * 32, E_DIM, lane);

  v8f acc[16] = {};
  float m = -1e30f, l = 0.0f;

  for (int k0 = 0; k0 < N_TOK; k0 += 32) {
    // async DMA of K[k0..k0+31][0..511] into LDS (issued by wave 0)
    if (wv == 0) {
      tdm_load_2d_f16(sK_off, kbp + (size_t)k0 * E_DIM,
                      /*tile*/ E_DIM, 32, /*tensor*/ E_DIM, 32, E_DIM);
      __builtin_amdgcn_s_wait_tensorcnt(0);
    }
    // prefetch next chunk of V while TDM runs
    if (k0 + 32 < N_TOK)
      __builtin_prefetch(vb + (size_t)(k0 + 32 + (threadIdx.x >> 3)) * E_DIM +
                             (threadIdx.x & 7) * 64, 0, 0);
    __syncthreads();

    // scores: 8 waves cover the 4x2 grid of 16x16 tiles (64 q x 32 keys)
    v8f s = {};
#pragma unroll
    for (int j = 0; j < 16; ++j) {
      v16h bb = load_b_h(sK + (size_t)(colPar * 16) * E_DIM + j * 32, E_DIM, lane);
      s = wmma16(qf[j], bb, s);
    }
#pragma unroll
    for (int i = 0; i < 8; ++i)
      sS[(rowTile * 16 + i + ((lane & 16) ? 8 : 0)) * 32 + colPar * 16 + (lane & 15)] = s[i];
    __syncthreads();

    if (threadIdx.x < 64) { // per-row online softmax stats
      const int row = threadIdx.x;
      float vals[32];
      float mc = -1e30f;
#pragma unroll
      for (int j = 0; j < 32; ++j) { vals[j] = sS[row * 32 + j]; mc = fmaxf(mc, vals[j]); }
      const float mnew = fmaxf(m, mc);
      const float fac = __expf(m - mnew);
      float ls = 0.0f;
#pragma unroll
      for (int j = 0; j < 32; ++j) {
        const float e = __expf(vals[j] - mnew);
        ls += e;
        sP[row * 32 + j] = (_Float16)e;
      }
      l = l * fac + ls;
      m = mnew;
      sFac[row] = fac;
    }
    __syncthreads();

    float fac_i[8];
#pragma unroll
    for (int i = 0; i < 8; ++i)
      fac_i[i] = sFac[rowTile * 16 + i + ((lane & 16) ? 8 : 0)];
    v16h a = load_a_h(sP + rowTile * 16 * 32, 32, lane);

    // P@V in two e-halves; wave owns e-tiles te = 2*j + colPar
#pragma unroll
    for (int pass = 0; pass < 2; ++pass) {
      { // stage V^T for e in [pass*256, pass*256+256)
        const int kv = threadIdx.x >> 3;
        const int e0 = (threadIdx.x & 7) * 32;
        const _Float16* vp = vb + (size_t)(k0 + kv) * E_DIM + pass * 256 + e0;
#pragma unroll
        for (int j8 = 0; j8 < 4; ++j8) {
          v8h vd = *(const v8h*)(vp + j8 * 8);
#pragma unroll
          for (int jj = 0; jj < 8; ++jj) sVt[(e0 + j8 * 8 + jj) * 32 + kv] = vd[jj];
        }
      }
      __syncthreads();
#pragma unroll
      for (int j = 0; j < 8; ++j) {
        const int jj = pass * 8 + j;               // accumulator index
        const int te = 2 * jj + colPar;            // global e-tile
#pragma unroll
        for (int i = 0; i < 8; ++i) acc[jj][i] *= fac_i[i];
        v16h bb = load_b_h(sVt + (te * 16 - pass * 256) * 32, 32, lane);
        acc[jj] = wmma16(a, bb, acc[jj]);
      }
      __syncthreads();
    }
  }

  if (threadIdx.x < 64) sInvL[threadIdx.x] = 1.0f / l;
  __syncthreads();
  float inv_i[8];
#pragma unroll
  for (int i = 0; i < 8; ++i)
    inv_i[i] = sInvL[rowTile * 16 + i + ((lane & 16) ? 8 : 0)];
  float* ob = out + (size_t)(b * N_TOK + q0 + rowTile * 16) * E_DIM;
#pragma unroll
  for (int jj = 0; jj < 16; ++jj) {
    const int col = (2 * jj + colPar) * 16 + (lane & 15);
#pragma unroll
    for (int i = 0; i < 8; ++i) {
      const int r = i + ((lane & 16) ? 8 : 0);
      ob[(size_t)r * E_DIM + col] = acc[jj][i] * inv_i[i];
    }
  }
}

// ---------------------------------------------------------------------------
// Kernel 3: S_col[b] = q^T @ k  (E x E, K = N), LDS-transposed chunk staging.
// ---------------------------------------------------------------------------
__global__ __launch_bounds__(256) void col_score_kernel(
    const _Float16* __restrict__ qh, const _Float16* __restrict__ kh,
    float* __restrict__ scol) {
  __shared__ alignas(32) _Float16 sQt[64 * 32]; // [d][n]
  __shared__ alignas(32) _Float16 sKt[64 * 32]; // [e][n]
  const int lane = threadIdx.x & 31;
  const int wv = threadIdx.x >> 5;
  const int rowTile = wv & 3;
  const int cPair = wv >> 2;
  const int b = blockIdx.z;
  const int dBase = blockIdx.y * 64;
  const int eBase = blockIdx.x * 64;
  const _Float16* qb = qh + (size_t)b * N_TOK * E_DIM;
  const _Float16* kbp = kh + (size_t)b * N_TOK * E_DIM;

  v8f acc[2] = {};
  for (int n0 = 0; n0 < N_TOK; n0 += 32) {
    {
      const int nl = threadIdx.x >> 3;
      const int d0 = (threadIdx.x & 7) * 8;
      v8h qd = *(const v8h*)(qb + (size_t)(n0 + nl) * E_DIM + dBase + d0);
      v8h kd = *(const v8h*)(kbp + (size_t)(n0 + nl) * E_DIM + eBase + d0);
#pragma unroll
      for (int j = 0; j < 8; ++j) {
        sQt[(d0 + j) * 32 + nl] = qd[j];
        sKt[(d0 + j) * 32 + nl] = kd[j];
      }
    }
    __syncthreads();
    v16h a = load_a_h(sQt + rowTile * 16 * 32, 32, lane);
#pragma unroll
    for (int t = 0; t < 2; ++t) {
      v16h bb = load_b_h(sKt + (cPair * 2 + t) * 16 * 32, 32, lane);
      acc[t] = wmma16(a, bb, acc[t]);
    }
    __syncthreads();
  }
  float* sp = scol + ((size_t)b * E_DIM + dBase + rowTile * 16) * E_DIM + eBase;
#pragma unroll
  for (int t = 0; t < 2; ++t) {
    const int col = (cPair * 2 + t) * 16 + (lane & 15);
#pragma unroll
    for (int i = 0; i < 8; ++i) {
      const int r = i + ((lane & 16) ? 8 : 0);
      sp[(size_t)r * E_DIM + col] = acc[t][i];
    }
  }
}

// ---------------------------------------------------------------------------
// Kernel 4: row-wise softmax of S_col -> P_col (f16). One wave per row.
// ---------------------------------------------------------------------------
__global__ __launch_bounds__(256) void col_softmax_kernel(
    const float* __restrict__ scol, _Float16* __restrict__ pcol) {
  const int lane = threadIdx.x & 31;
  const int wv = threadIdx.x >> 5;
  const int row = blockIdx.x * 8 + wv;
  const float* sp = scol + (size_t)row * E_DIM;
  float vals[16];
  float m = -1e30f;
#pragma unroll
  for (int j = 0; j < 16; ++j) {
    vals[j] = sp[lane + j * 32];
    m = fmaxf(m, vals[j]);
  }
#pragma unroll
  for (int off = 16; off >= 1; off >>= 1) m = fmaxf(m, __shfl_xor(m, off, 32));
  float l = 0.0f;
#pragma unroll
  for (int j = 0; j < 16; ++j) { vals[j] = __expf(vals[j] - m); l += vals[j]; }
#pragma unroll
  for (int off = 16; off >= 1; off >>= 1) l += __shfl_xor(l, off, 32);
  const float inv = 1.0f / l;
  _Float16* pp = pcol + (size_t)row * E_DIM;
#pragma unroll
  for (int j = 0; j < 16; ++j) pp[lane + j * 32] = (_Float16)(vals[j] * inv);
}

// ---------------------------------------------------------------------------
// Kernel 5: out += v @ P_col^T  (P row-major is already K-contiguous for B).
// ---------------------------------------------------------------------------
__global__ __launch_bounds__(256) void col_apply_kernel(
    const _Float16* __restrict__ vh, const _Float16* __restrict__ pcol,
    float* __restrict__ out) {
  const int lane = threadIdx.x & 31;
  const int wv = threadIdx.x >> 5;
  const int b = blockIdx.z;
  const int nBase = blockIdx.y * 128 + wv * 16;
  const int dBase = blockIdx.x * 64;
  const _Float16* vb = vh + (size_t)(b * N_TOK + nBase) * E_DIM;
  const _Float16* pb = pcol + (size_t)b * E_DIM * E_DIM;

  v8f acc[4] = {};
  for (int e0 = 0; e0 < E_DIM; e0 += 32) {
    v16h a = load_a_h(vb + e0, E_DIM, lane);
#pragma unroll
    for (int ct = 0; ct < 4; ++ct) {
      const int d = dBase + ct * 16 + (lane & 15);
      const int kb = (lane >> 4) << 4;
      v16h bb = *(const v16h*)(pb + (size_t)d * E_DIM + e0 + kb);
      acc[ct] = wmma16(a, bb, acc[ct]);
    }
  }
  float* ob = out + (size_t)(b * N_TOK + nBase) * E_DIM;
#pragma unroll
  for (int ct = 0; ct < 4; ++ct) {
    const int col = dBase + ct * 16 + (lane & 15);
#pragma unroll
    for (int i = 0; i < 8; ++i) {
      const int r = i + ((lane & 16) ? 8 : 0);
      ob[(size_t)r * E_DIM + col] += acc[ct][i];
    }
  }
}

extern "C" void kernel_launch(void* const* d_in, const int* in_sizes, int n_in,
                              void* d_out, int out_size, void* d_ws, size_t ws_size,
                              hipStream_t stream) {
  (void)in_sizes; (void)n_in; (void)out_size; (void)ws_size;
  const float* x = (const float*)d_in[0];
  const float* w = (const float*)d_in[1];
  const float* bias = (const float*)d_in[2];
  float* out = (float*)d_out;
  char* ws = (char*)d_ws;

  const size_t SEG = (size_t)BATCH * N_TOK * E_DIM * sizeof(_Float16); // 64 MB
  _Float16* qh = (_Float16*)(ws);
  _Float16* kh = (_Float16*)(ws + SEG);
  _Float16* vh = (_Float16*)(ws + 2 * SEG);
  float* scol = (float*)(ws + 3 * SEG);                                // 32 MB
  _Float16* pcol =
      (_Float16*)(ws + 3 * SEG + (size_t)BATCH * E_DIM * E_DIM * sizeof(float));

  qkv_gemm_kernel<<<dim3(512, 24), 256, 0, stream>>>(x, w, bias, qh, kh, vh);
  row_attn_kernel<<<dim3(32, 32), 256, 0, stream>>>(qh, kh, vh, out);
  col_score_kernel<<<dim3(8, 8, 32), 256, 0, stream>>>(qh, kh, scol);
  col_softmax_kernel<<<2048, 256, 0, stream>>>(scol, pcol);
  col_apply_kernel<<<dim3(8, 16, 32), 256, 0, stream>>>(vh, pcol, out);
}
